// DecoderAttentionBlock_73203422593722
// MI455X (gfx1250) — compile-verified, matrix operand order checked by reading, not run on hardware
//
#include <hip/hip_runtime.h>
#include <hip/hip_bf16.h>

typedef _Float16 h16;
typedef __attribute__((ext_vector_type(16))) _Float16 v16h;
typedef __attribute__((ext_vector_type(8)))  _Float16 v8h;
typedef __attribute__((ext_vector_type(4)))  _Float16 v4h;
typedef __attribute__((ext_vector_type(8)))  float    v8f;

#define DM   1024
#define NH   16
#define HD   64
#define TQ   512
#define SK   1024
#define BB   16

// ---------------- WMMA helpers ----------------
// A-fragment (16x32, f16): lane L holds row L%16; lanes 0-15 carry K=k0+{0..7,16..23},
// lanes 16-31 carry K=k0+{8..15,24..31}. B-fragment of W[N,K] row-major is identical
// code with row = output column. Two 16B loads per lane per fragment.
__device__ __forceinline__ v16h load_frag16(const h16* __restrict__ base, int ld, int k0, int lane) {
  const int row = lane & 15;
  const int kb  = k0 + ((lane & 16) ? 8 : 0);
  const h16* p = base + (size_t)row * ld + kb;
  v8h lo = *(const v8h*)(p);
  v8h hi = *(const v8h*)(p + 16);
  v16h r;
#pragma unroll
  for (int i = 0; i < 8; ++i) { r[i] = lo[i]; r[i + 8] = hi[i]; }
  return r;
}

__device__ __forceinline__ v8f wmma_f16(v16h a, v16h b, v8f c) {
  return __builtin_amdgcn_wmma_f32_16x16x32_f16(false, a, false, b, (short)0, c, false, false);
}

// max across the 16 lanes of a DPP row via ROW_XMASK (wave32: row == 16 lanes).
// Pure VALU; avoids ds_bpermute + s_wait_dscnt in the flash inner loop.
template <int MASK>
__device__ __forceinline__ float xmask_max(float v) {
  int i = __float_as_int(v);
  int j = __builtin_amdgcn_update_dpp(i, i, 0x160 | MASK, 0xf, 0xf, false);
  return fmaxf(v, __int_as_float(j));
}
__device__ __forceinline__ float redmax16(float v) {
  v = xmask_max<1>(v);
  v = xmask_max<2>(v);
  v = xmask_max<4>(v);
  v = xmask_max<8>(v);
  return v;
}

// ---------------- LayerNorm (fp32 math) -> f16 ----------------
__global__ void ln_f16_kernel(const float* __restrict__ x, const float* __restrict__ g,
                              const float* __restrict__ b, h16* __restrict__ y) {
  const int row = blockIdx.x;
  const int t = threadIdx.x;              // 256 threads * 4 cols = 1024
  const float4 v = ((const float4*)(x + (size_t)row * DM))[t];
  float s  = v.x + v.y + v.z + v.w;
  float s2 = v.x*v.x + v.y*v.y + v.z*v.z + v.w*v.w;
  __shared__ float rs[256], rq[256];
  rs[t] = s; rq[t] = s2;
  __syncthreads();
  for (int o = 128; o > 0; o >>= 1) {
    if (t < o) { rs[t] += rs[t + o]; rq[t] += rq[t + o]; }
    __syncthreads();
  }
  const float mean = rs[0] * (1.0f / DM);
  const float var  = rq[0] * (1.0f / DM) - mean * mean;
  const float rinv = rsqrtf(var + 1e-5f);
  const float4 gv = ((const float4*)g)[t];
  const float4 bv = ((const float4*)b)[t];
  v4h o;
  o[0] = (h16)((v.x - mean) * rinv * gv.x + bv.x);
  o[1] = (h16)((v.y - mean) * rinv * gv.y + bv.y);
  o[2] = (h16)((v.z - mean) * rinv * gv.z + bv.z);
  o[3] = (h16)((v.w - mean) * rinv * gv.w + bv.w);
  ((v4h*)(y + (size_t)row * DM))[t] = o;
}

// ---------------- f32 -> f16 convert ----------------
__global__ void cvt_f16_kernel(const float* __restrict__ src, h16* __restrict__ dst, int n4) {
  int i = blockIdx.x * blockDim.x + threadIdx.x;
  if (i < n4) {
    float4 v = ((const float4*)src)[i];
    v4h o; o[0] = (h16)v.x; o[1] = (h16)v.y; o[2] = (h16)v.z; o[3] = (h16)v.w;
    ((v4h*)dst)[i] = o;
  }
}

// ---------------- head permutes ----------------
// src: [T*B rows (t*B+b), srcStride] f16. dst row-major [bh][T][64] or transposed [bh][64][T].
__global__ void permute_heads_kernel(const h16* __restrict__ src, h16* __restrict__ dst,
                                     int T, int B, int srcStride, int col0, int transposed, int n) {
  int i = blockIdx.x * blockDim.x + threadIdx.x;
  if (i >= n) return;
  int d  = i & 63;
  int h  = (i >> 6) & 15;
  int tb = i >> 10;
  int b  = tb % B;
  int t  = tb / B;
  h16 v = src[(size_t)tb * srcStride + col0 + h * HD + d];
  int bh = b * NH + h;
  if (!transposed) dst[((size_t)bh * T + t) * HD + d] = v;
  else             dst[((size_t)bh * HD + d) * T + t] = v;
}

// ---------------- WMMA GEMM: out = A[MxK] * W[NxK]^T + bias (+epilogue) ----------------
// Wave tile 32(M) x 64(N): 2 A-frags + 4 B-frags -> 8 WMMAs per K-step.
// EPI: 0 = f16 out, 1 = f32 out + residual add, 2 = f16 out with quickGELU
template <int EPI>
__global__ void gemm_wmma_kernel(const h16* __restrict__ A, const h16* __restrict__ W,
                                 const float* __restrict__ bias, const float* __restrict__ res,
                                 void* __restrict__ outp, int M, int N, int K) {
  const int lane  = threadIdx.x & 31;
  const int wtile = blockIdx.x * (blockDim.x >> 5) + (threadIdx.x >> 5);
  const int tn    = N >> 6;
  const int m0    = (wtile / tn) << 5;      // 32 rows
  const int n0    = (wtile % tn) << 6;      // 64 cols
  const h16* Ab0 = A + (size_t)m0 * K;
  const h16* Ab1 = A + (size_t)(m0 + 16) * K;
  const h16* Wb  = W + (size_t)n0 * K;
  v8f acc[2][4];
#pragma unroll
  for (int i = 0; i < 2; ++i)
#pragma unroll
    for (int j = 0; j < 4; ++j) acc[i][j] = (v8f){};

  for (int k = 0; k < K; k += 32) {
    v16h a0 = load_frag16(Ab0, K, k, lane);
    v16h a1 = load_frag16(Ab1, K, k, lane);
    v16h b0 = load_frag16(Wb,                  K, k, lane);
    v16h b1 = load_frag16(Wb + 16 * (size_t)K, K, k, lane);
    v16h b2 = load_frag16(Wb + 32 * (size_t)K, K, k, lane);
    v16h b3 = load_frag16(Wb + 48 * (size_t)K, K, k, lane);
    acc[0][0] = wmma_f16(a0, b0, acc[0][0]);
    acc[0][1] = wmma_f16(a0, b1, acc[0][1]);
    acc[0][2] = wmma_f16(a0, b2, acc[0][2]);
    acc[0][3] = wmma_f16(a0, b3, acc[0][3]);
    acc[1][0] = wmma_f16(a1, b0, acc[1][0]);
    acc[1][1] = wmma_f16(a1, b1, acc[1][1]);
    acc[1][2] = wmma_f16(a1, b2, acc[1][2]);
    acc[1][3] = wmma_f16(a1, b3, acc[1][3]);
  }
  const int rowoff = (lane & 16) ? 8 : 0;
  const int nc     = lane & 15;
#pragma unroll
  for (int i = 0; i < 2; ++i) {
    const int mb = m0 + 16 * i + rowoff;
#pragma unroll
    for (int j = 0; j < 4; ++j) {
      const int n = n0 + 16 * j + nc;
      const float bv = bias[n];
#pragma unroll
      for (int r = 0; r < 8; ++r) {
        const size_t idx = (size_t)(mb + r) * N + n;
        float v = acc[i][j][r] + bv;
        if (EPI == 1) {
          ((float*)outp)[idx] = v + res[idx];
        } else if (EPI == 2) {
          float gg = v / (1.0f + __expf(-1.702f * v));   // quickGELU
          ((h16*)outp)[idx] = (h16)gg;
        } else {
          ((h16*)outp)[idx] = (h16)v;
        }
      }
    }
  }
}

// ---------------- flash attention (one 16-row Q tile per wave) ----------------
// Qh: [bh][T][64], Kh: [bh][S][64], Vt: [bh][64][S]; ctx: [T*B, 1024] f16.
// Row-max via DPP ROW_XMASK; row-sum via WMMA against an all-ones B fragment.
__global__ void attn_wmma_kernel(const h16* __restrict__ Qh, const h16* __restrict__ Kh,
                                 const h16* __restrict__ Vt, h16* __restrict__ ctx,
                                 int T, int S, float scale) {
  __shared__ h16 pbuf[8][16 * 32];
  const int lane = threadIdx.x & 31;
  const int wave = threadIdx.x >> 5;
  const int gw   = blockIdx.x * 8 + wave;
  const int qt   = T >> 4;
  const int bh   = gw / qt;
  const int t0   = (gw % qt) << 4;
  const h16* Qb = Qh + ((size_t)bh * T + t0) * HD;
  const h16* Kb = Kh + (size_t)bh * S * HD;
  const h16* Vb = Vt + (size_t)bh * HD * S;

  const v16h qa0 = load_frag16(Qb, HD, 0,  lane);   // reused every iteration
  const v16h qa1 = load_frag16(Qb, HD, 32, lane);
  v16h ones;
#pragma unroll
  for (int i = 0; i < 16; ++i) ones[i] = (h16)1.0f;

  v8f o0 = {}, o1 = {}, o2 = {}, o3 = {};
  float mrow[8], lrow[8];
#pragma unroll
  for (int r = 0; r < 8; ++r) { mrow[r] = -3.0e38f; lrow[r] = 0.0f; }

  const int rowoff = (lane & 16) ? 8 : 0;
  const int col    = lane & 15;
  h16* P = &pbuf[wave][0];

  for (int s0 = 0; s0 < S; s0 += 32) {
    // scores: two 16x16 tiles over 32 keys, K-dim 64 = 2 WMMAs each
    v8f sc0 = {}, sc1 = {};
    {
      v16h kb0 = load_frag16(Kb + (size_t)s0 * HD,        HD, 0,  lane);
      v16h kb1 = load_frag16(Kb + (size_t)s0 * HD,        HD, 32, lane);
      sc0 = wmma_f16(qa0, kb0, sc0);
      sc0 = wmma_f16(qa1, kb1, sc0);
      v16h kb2 = load_frag16(Kb + (size_t)(s0 + 16) * HD, HD, 0,  lane);
      v16h kb3 = load_frag16(Kb + (size_t)(s0 + 16) * HD, HD, 32, lane);
      sc1 = wmma_f16(qa0, kb2, sc1);
      sc1 = wmma_f16(qa1, kb3, sc1);
    }
    // online softmax update (fp32); stage P into LDS as f16
    float al[8];
#pragma unroll
    for (int r = 0; r < 8; ++r) {
      float a0 = sc0[r] * scale, a1 = sc1[r] * scale;
      float mx = redmax16(fmaxf(a0, a1));           // DPP, no LDS
      float mn = fmaxf(mrow[r], mx);
      al[r] = __expf(mrow[r] - mn);
      mrow[r] = mn;
      float p0 = __expf(a0 - mn), p1 = __expf(a1 - mn);
      o0[r] *= al[r]; o1[r] *= al[r]; o2[r] *= al[r]; o3[r] *= al[r];
      P[(r + rowoff) * 32 + col]      = (h16)p0;
      P[(r + rowoff) * 32 + 16 + col] = (h16)p1;
    }
    asm volatile("s_wait_dscnt 0" ::: "memory");    // P store -> P load (same wave, LDS)
    v16h pa = load_frag16(P, 32, 0, lane);          // P as 16x32 A-fragment
    v8f rsum = wmma_f16(pa, ones, (v8f){});         // row sums of P, broadcast over lanes
#pragma unroll
    for (int r = 0; r < 8; ++r) lrow[r] = lrow[r] * al[r] + rsum[r];
    {
      v16h vb0 = load_frag16(Vb + 0  * (size_t)S, S, s0, lane);
      o0 = wmma_f16(pa, vb0, o0);
      v16h vb1 = load_frag16(Vb + 16 * (size_t)S, S, s0, lane);
      o1 = wmma_f16(pa, vb1, o1);
      v16h vb2 = load_frag16(Vb + 32 * (size_t)S, S, s0, lane);
      o2 = wmma_f16(pa, vb2, o2);
      v16h vb3 = load_frag16(Vb + 48 * (size_t)S, S, s0, lane);
      o3 = wmma_f16(pa, vb3, o3);
    }
  }
  // write ctx rows back into [t*B+b, h*64+d] layout for the out-proj GEMM
  const int b = bh / NH, h = bh % NH;
#pragma unroll
  for (int r = 0; r < 8; ++r) {
    const float inv = 1.0f / lrow[r];
    const int t = t0 + r + rowoff;
    const size_t base = ((size_t)t * BB + b) * DM + h * HD;
    ctx[base + 0  + col] = (h16)(o0[r] * inv);
    ctx[base + 16 + col] = (h16)(o1[r] * inv);
    ctx[base + 32 + col] = (h16)(o2[r] * inv);
    ctx[base + 48 + col] = (h16)(o3[r] * inv);
  }
}

// ---------------- host orchestration ----------------
extern "C" void kernel_launch(void* const* d_in, const int* in_sizes, int n_in,
                              void* d_out, int out_size, void* d_ws, size_t ws_size,
                              hipStream_t stream) {
  (void)in_sizes; (void)n_in; (void)out_size; (void)ws_size;
  const float* x        = (const float*)d_in[0];
  const float* hidden   = (const float*)d_in[1];
  const float* ln1_g    = (const float*)d_in[2];
  const float* ln1_b    = (const float*)d_in[3];
  const float* sa_in_w  = (const float*)d_in[4];
  const float* sa_in_b  = (const float*)d_in[5];
  const float* sa_out_w = (const float*)d_in[6];
  const float* sa_out_b = (const float*)d_in[7];
  const float* ln2_g    = (const float*)d_in[8];
  const float* ln2_b    = (const float*)d_in[9];
  const float* ca_in_w  = (const float*)d_in[10];
  const float* ca_in_b  = (const float*)d_in[11];
  const float* ca_out_w = (const float*)d_in[12];
  const float* ca_out_b = (const float*)d_in[13];
  const float* ln3_g    = (const float*)d_in[14];
  const float* ln3_b    = (const float*)d_in[15];
  const float* fc_w     = (const float*)d_in[16];
  const float* fc_b     = (const float*)d_in[17];
  const float* proj_w   = (const float*)d_in[18];
  const float* proj_b   = (const float*)d_in[19];

  const size_t MB = (size_t)1 << 20;
  char* ws = (char*)d_ws;
  float* x1   = (float*)(ws + 0);          // 32 MB: residual stream after self-attn
  h16* actA   = (h16*)(ws + 32 * MB);      // 64 MB: ln outputs / hidden_f16 / q_ca
  h16* gemmO  = (h16*)(ws + 96 * MB);      // 64 MB: qkv_sa / kv_ca / mlp hidden
  h16* QhB    = (h16*)(ws + 160 * MB);     // 16 MB
  h16* KhB    = (h16*)(ws + 176 * MB);     // 32 MB
  h16* VtB    = (h16*)(ws + 208 * MB);     // 32 MB
  h16* ctx    = (h16*)(ws + 240 * MB);     // 16 MB
  h16* wf16   = (h16*)(ws + 256 * MB);     // 32 MB: all weights as f16
  h16* w_sa_in  = wf16;
  h16* w_sa_out = wf16 + (size_t)3  * 1024 * 1024;
  h16* w_ca_in  = wf16 + (size_t)4  * 1024 * 1024;
  h16* w_ca_out = wf16 + (size_t)7  * 1024 * 1024;
  h16* w_fc     = wf16 + (size_t)8  * 1024 * 1024;
  h16* w_proj   = wf16 + (size_t)12 * 1024 * 1024;
  float* xo = (float*)d_out;               // residual stream after cross-attn = final output

  const int M  = TQ * BB;                  // 8192 query tokens
  const int MS = SK * BB;                  // 16384 kv tokens
  const float scale = 0.125f;              // 1/sqrt(64)

  auto cvt = [&](const float* s, h16* d, size_t n) {
    int n4 = (int)(n / 4);
    hipLaunchKernelGGL(cvt_f16_kernel, dim3((n4 + 255) / 256), dim3(256), 0, stream, s, d, n4);
  };
  cvt(sa_in_w,  w_sa_in,  (size_t)3 * DM * DM);
  cvt(sa_out_w, w_sa_out, (size_t)DM * DM);
  cvt(ca_in_w,  w_ca_in,  (size_t)3 * DM * DM);
  cvt(ca_out_w, w_ca_out, (size_t)DM * DM);
  cvt(fc_w,     w_fc,     (size_t)4 * DM * DM);
  cvt(proj_w,   w_proj,   (size_t)4 * DM * DM);

  // ---- phase A: self attention ----
  hipLaunchKernelGGL(ln_f16_kernel, dim3(M), dim3(256), 0, stream, x, ln1_g, ln1_b, actA);
  hipLaunchKernelGGL((gemm_wmma_kernel<0>), dim3((M / 32) * (3 * DM / 64) / 8), dim3(256), 0, stream,
                     actA, w_sa_in, sa_in_b, (const float*)nullptr, (void*)gemmO, M, 3 * DM, DM);
  {
    int n = M * DM;
    hipLaunchKernelGGL(permute_heads_kernel, dim3(n / 256), dim3(256), 0, stream, gemmO, QhB, TQ, BB, 3 * DM, 0,      0, n);
    hipLaunchKernelGGL(permute_heads_kernel, dim3(n / 256), dim3(256), 0, stream, gemmO, KhB, TQ, BB, 3 * DM, DM,     0, n);
    hipLaunchKernelGGL(permute_heads_kernel, dim3(n / 256), dim3(256), 0, stream, gemmO, VtB, TQ, BB, 3 * DM, 2 * DM, 1, n);
  }
  hipLaunchKernelGGL(attn_wmma_kernel, dim3((BB * NH) * (TQ / 16) / 8), dim3(256), 0, stream,
                     QhB, KhB, VtB, ctx, TQ, TQ, scale);
  hipLaunchKernelGGL((gemm_wmma_kernel<1>), dim3((M / 32) * (DM / 64) / 8), dim3(256), 0, stream,
                     ctx, w_sa_out, sa_out_b, x, (void*)x1, M, DM, DM);

  // ---- phase B: cross attention ----
  h16* xln2 = actA;
  h16* hidf = actA + (size_t)8  * 1024 * 1024;   // 32 MB region
  h16* qca  = actA + (size_t)24 * 1024 * 1024;   // 16 MB region
  hipLaunchKernelGGL(ln_f16_kernel, dim3(M), dim3(256), 0, stream, x1, ln2_g, ln2_b, xln2);
  cvt(hidden, hidf, (size_t)MS * DM);
  hipLaunchKernelGGL((gemm_wmma_kernel<0>), dim3((M / 32) * (DM / 64) / 8), dim3(256), 0, stream,
                     xln2, w_ca_in, ca_in_b, (const float*)nullptr, (void*)qca, M, DM, DM);
  hipLaunchKernelGGL((gemm_wmma_kernel<0>), dim3((MS / 32) * (2 * DM / 64) / 8), dim3(256), 0, stream,
                     hidf, w_ca_in + (size_t)DM * DM, ca_in_b + DM, (const float*)nullptr,
                     (void*)gemmO, MS, 2 * DM, DM);
  {
    int nq = M * DM, nk = MS * DM;
    hipLaunchKernelGGL(permute_heads_kernel, dim3(nq / 256), dim3(256), 0, stream, qca,   QhB, TQ, BB, DM,     0,  0, nq);
    hipLaunchKernelGGL(permute_heads_kernel, dim3(nk / 256), dim3(256), 0, stream, gemmO, KhB, SK, BB, 2 * DM, 0,  0, nk);
    hipLaunchKernelGGL(permute_heads_kernel, dim3(nk / 256), dim3(256), 0, stream, gemmO, VtB, SK, BB, 2 * DM, DM, 1, nk);
  }
  hipLaunchKernelGGL(attn_wmma_kernel, dim3((BB * NH) * (TQ / 16) / 8), dim3(256), 0, stream,
                     QhB, KhB, VtB, ctx, TQ, SK, scale);
  hipLaunchKernelGGL((gemm_wmma_kernel<1>), dim3((M / 32) * (DM / 64) / 8), dim3(256), 0, stream,
                     ctx, w_ca_out, ca_out_b, x1, (void*)xo, M, DM, DM);

  // ---- phase C: MLP ----
  h16* xln3 = actA;
  hipLaunchKernelGGL(ln_f16_kernel, dim3(M), dim3(256), 0, stream, xo, ln3_g, ln3_b, xln3);
  hipLaunchKernelGGL((gemm_wmma_kernel<2>), dim3((M / 32) * (4 * DM / 64) / 8), dim3(256), 0, stream,
                     xln3, w_fc, fc_b, (const float*)nullptr, (void*)gemmO, M, 4 * DM, DM);
  hipLaunchKernelGGL((gemm_wmma_kernel<1>), dim3((M / 32) * (DM / 64) / 8), dim3(256), 0, stream,
                     gemmO, w_proj, proj_b, xo, (void*)xo, M, DM, 4 * DM);
}
